// MyViT_76441827934333
// MI455X (gfx1250) — matrix operand order repaired and checked by reading, not compiled, not run on hardware
//
#include <hip/hip_runtime.h>
#include <hip/hip_bf16.h>
#include <math.h>

#define BATCH 64
#define SEQQ 197
#define HID 768
#define NH 12
#define DH 64
#define NP 196
#define OUTD 1000
#define IMGW 224
#define PSZ 16
#define NPAT 14
#define TPAD 224   // padded t-dimension for P@V (multiple of 32, >= SEQQ)

typedef __attribute__((ext_vector_type(16))) _Float16 v16h;
typedef __attribute__((ext_vector_type(8)))  _Float16 v8h;
typedef __attribute__((ext_vector_type(8)))  float    v8f;

__device__ __forceinline__ int imin(int a, int b){ return a < b ? a : b; }

// ---- WMMA fragment helpers (CDNA5 ISA 7.12.2 layouts, wave32) ----
// 16-bit A (16x32): lane's data = Arow[8h..8h+7] ++ Arow[16+8h..16+8h+7]  (two b128 loads)
__device__ __forceinline__ v16h load_a_contig(const _Float16* Arow, int kbase, int half) {
  v8h lo = *(const v8h*)(Arow + kbase + 8*half);
  v8h hi = *(const v8h*)(Arow + kbase + 16 + 8*half);
  return __builtin_shufflevector(lo, hi, 0,1,2,3,4,5,6,7,8,9,10,11,12,13,14,15);
}

// A from f32 source (head GEMM): same contiguous pattern, converted
__device__ __forceinline__ v16h load_a32_contig(const float* Arow, int kbase, int half) {
  const float* p0 = Arow + kbase + 8*half;
  const float* p1 = Arow + kbase + 16 + 8*half;
  v16h a;
#pragma unroll
  for (int i = 0; i < 8; ++i) { a[i] = (_Float16)p0[i]; a[i+8] = (_Float16)p1[i]; }
  return a;
}

// 16-bit B (32x16): lane holds column; with K-major (transposed) storage the lane's
// data = Brow[kbase+8h .. kbase+8h+7] ++ Brow[kbase+16+8h .. kbase+16+8h+7]  (two b128 loads)
__device__ __forceinline__ v16h load_b_contig(const _Float16* Brow, int kbase, int half) {
  v8h lo = *(const v8h*)(Brow + kbase + 8*half);
  v8h hi = *(const v8h*)(Brow + kbase + 16 + 8*half);
  return __builtin_shufflevector(lo, hi, 0,1,2,3,4,5,6,7,8,9,10,11,12,13,14,15);
}

__device__ __forceinline__ v8f wmma_f16(v16h a, v16h b, v8f c) {
  return __builtin_amdgcn_wmma_f32_16x16x32_f16(false, a, false, b, (short)0, c, false, false);
}

// ---------------- elementwise / setup kernels ----------------
__global__ void k_pos(float* __restrict__ pe) {
  int idx = blockIdx.x*blockDim.x + threadIdx.x;
  if (idx >= SEQQ*HID) return;
  int s = idx / HID, d = idx % HID;
  float fd = (float)((d & 1) == 0 ? d : (d - 1));
  float ang = (float)s * __powf(10000.0f, -fd / (float)HID);
  pe[idx] = ((d & 1) == 0) ? __sinf(ang) : __cosf(ang);
}

// transpose-convert: dst[n*K + k] = (f16)src[k*N + n]
__global__ void k_cvtT(const float* __restrict__ s, _Float16* __restrict__ d, int K, int N) {
  int idx = blockIdx.x*blockDim.x + threadIdx.x;
  if (idx >= K*N) return;
  int n = idx / K, k = idx % K;
  d[idx] = (_Float16)s[(long)k*N + n];
}

// per-head transpose-convert: dst[h][e][d] = (f16)src[h][d][e]
__global__ void k_cvtT_heads(const float* __restrict__ s, _Float16* __restrict__ d) {
  int idx = blockIdx.x*blockDim.x + threadIdx.x;
  if (idx >= NH*DH*DH) return;
  int h = idx / (DH*DH), r = idx % (DH*DH);
  int e = r / DH, dd = r % DH;
  d[idx] = (_Float16)s[(h*DH + dd)*DH + e];
}

__global__ void k_patches(const float* __restrict__ img, _Float16* __restrict__ pt) {
  int idx = blockIdx.x*blockDim.x + threadIdx.x;
  if (idx >= BATCH*NP*HID) return;
  int n = idx / (NP*HID);
  int r = idx % (NP*HID);
  int p = r / HID, k = r % HID;
  int py = p / NPAT, px = p % NPAT;
  int c = k >> 8, rem = k & 255, ii = rem >> 4, jj = rem & 15;
  pt[idx] = (_Float16)img[(((long)(n*3 + c)*IMGW) + py*PSZ + ii)*IMGW + px*PSZ + jj];
}

__global__ void k_cls(const float* __restrict__ cls, const float* __restrict__ pe,
                      float* __restrict__ tokens) {
  int idx = blockIdx.x*blockDim.x + threadIdx.x;
  if (idx >= BATCH*HID) return;
  int n = idx / HID, d = idx % HID;
  tokens[(long)n*SEQQ*HID + d] = cls[d] + pe[d];
}

__global__ void k_stats(const float* __restrict__ x, float* __restrict__ mu, float* __restrict__ rstd) {
  __shared__ float sh[256], sh2[256];
  int b = blockIdx.x;
  const float* p = x + (long)b*SEQQ*HID;
  float s = 0.f, s2 = 0.f;
  for (int i = threadIdx.x; i < SEQQ*HID; i += blockDim.x) { float v = p[i]; s += v; s2 += v*v; }
  sh[threadIdx.x] = s; sh2[threadIdx.x] = s2; __syncthreads();
  for (int st = 128; st > 0; st >>= 1) {
    if ((int)threadIdx.x < st) { sh[threadIdx.x] += sh[threadIdx.x + st]; sh2[threadIdx.x] += sh2[threadIdx.x + st]; }
    __syncthreads();
  }
  if (threadIdx.x == 0) {
    float inv_n = 1.0f / (float)(SEQQ*HID);
    float m = sh[0] * inv_n;
    float var = sh2[0] * inv_n - m*m;
    mu[b] = m; rstd[b] = rsqrtf(var + 1e-5f);
  }
}

__global__ void k_ln(const float* __restrict__ x, const float* __restrict__ mu,
                     const float* __restrict__ rstd, const float* __restrict__ g,
                     const float* __restrict__ beta, _Float16* __restrict__ o) {
  long idx = (long)blockIdx.x*blockDim.x + threadIdx.x;
  if (idx >= (long)BATCH*SEQQ*HID) return;
  int b = (int)(idx / (SEQQ*HID));
  int r = (int)(idx % (SEQQ*HID));
  o[idx] = (_Float16)((x[idx] - mu[b]) * rstd[b] * g[r] + beta[r]);
}

// zero the padded tail of Vt: t in [SEQQ, TPAD) for every (b,h,e)
__global__ void k_vpad(_Float16* __restrict__ vt) {
  int idx = blockIdx.x*blockDim.x + threadIdx.x;
  const int per = TPAD - SEQQ;
  if (idx >= BATCH*NH*DH*per) return;
  int c = idx / per, t = SEQQ + idx % per;
  vt[(long)c*TPAD + t] = (_Float16)0.f;
}

// ---------------- WMMA GEMM kernels ----------------
// tokens[b, 1+p, :] = patches[b,p,:] @ Wmap + b_map + pe[1+p,:]   (WmapT: [n][k])
__global__ void __launch_bounds__(128) k_embed(const _Float16* __restrict__ A16,
    const _Float16* __restrict__ WT, const float* __restrict__ bias,
    const float* __restrict__ pe, float* __restrict__ tokens) {
  int b = blockIdx.x, mt = blockIdx.y, nt = blockIdx.z;
  int lane = threadIdx.x & 31, wave = threadIdx.x >> 5;
  int half = lane >> 4, ml = lane & 15;
  int rowc = imin(mt*16 + ml, NP - 1);
  const _Float16* Arow = A16 + ((long)b*NP + rowc)*HID;
  int n0 = nt*64 + wave*16, col = n0 + ml;
  const _Float16* Brow = WT + (long)col*HID;
  v8f acc = {};
  for (int k = 0; k < HID; k += 32)
    acc = wmma_f16(load_a_contig(Arow, k, half), load_b_contig(Brow, k, half), acc);
#pragma unroll
  for (int r = 0; r < 8; ++r) {
    int m = mt*16 + r + 8*half;
    if (m < NP) {
      int s = m + 1;
      long o = ((long)b*SEQQ + s)*HID + col;
      tokens[o] = acc[r] + bias[col] + pe[s*HID + col];
    }
  }
}

// q/k[b,h,s,e], vT[b,h,e,t] from x16 @ W{q,k,v}T + bias
__global__ void __launch_bounds__(128) k_qkv(const _Float16* __restrict__ x16,
    const _Float16* __restrict__ wqT, const _Float16* __restrict__ wkT, const _Float16* __restrict__ wvT,
    const float* __restrict__ bq, const float* __restrict__ bk, const float* __restrict__ bv,
    _Float16* __restrict__ q, _Float16* __restrict__ ko, _Float16* __restrict__ vt) {
  int b = blockIdx.x, h = blockIdx.y, mt = blockIdx.z;
  int lane = threadIdx.x & 31, wave = threadIdx.x >> 5;
  int half = lane >> 4, ml = lane & 15;
  int rowc = imin(mt*16 + ml, SEQQ - 1);
  const _Float16* Arow = x16 + ((long)b*SEQQ + rowc)*HID + h*DH;
  v16h a0 = load_a_contig(Arow, 0, half);
  v16h a1 = load_a_contig(Arow, 32, half);
  int e0 = wave*16, col = e0 + ml;
  const _Float16* Ws[3] = { wqT + h*DH*DH, wkT + h*DH*DH, wvT + h*DH*DH };
  const float*    Bs[3] = { bq + h*DH, bk + h*DH, bv + h*DH };
#pragma unroll
  for (int t = 0; t < 3; ++t) {
    const _Float16* Brow = Ws[t] + (long)col*DH;   // transposed: [e][d]
    v8f acc = {};
    acc = wmma_f16(a0, load_b_contig(Brow, 0,  half), acc);
    acc = wmma_f16(a1, load_b_contig(Brow, 32, half), acc);
    float bb = Bs[t][col];
    if (t < 2) {
      _Float16* D = (t == 0 ? q : ko) + ((long)b*NH + h)*SEQQ*DH;
#pragma unroll
      for (int r = 0; r < 8; ++r) {
        int m = mt*16 + r + 8*half;
        if (m < SEQQ) D[(long)m*DH + col] = (_Float16)(acc[r] + bb);
      }
    } else {
      _Float16* D = vt + (((long)b*NH + h)*DH + col)*TPAD;   // transposed: [e][t]
#pragma unroll
      for (int r = 0; r < 8; ++r) {
        int m = mt*16 + r + 8*half;
        if (m < SEQQ) D[m] = (_Float16)(acc[r] + bb);
      }
    }
  }
}

// attention for one (b, h, 16-query tile): scores in LDS, softmax, P@V, residual add
__global__ void __launch_bounds__(128) k_attn(const _Float16* __restrict__ q,
    const _Float16* __restrict__ kk, const _Float16* __restrict__ vt,
    const float* __restrict__ tokens, float* __restrict__ outb) {
  __shared__ float    S[16][TPAD];
  __shared__ __align__(16) _Float16 P[16][TPAD];
  int b = blockIdx.x, h = blockIdx.y, mt = blockIdx.z;
  int lane = threadIdx.x & 31, wave = threadIdx.x >> 5;
  int half = lane >> 4, ml = lane & 15;
  long base = ((long)b*NH + h)*SEQQ*DH;
  const _Float16* Q = q + base;
  const _Float16* K = kk + base;
  const _Float16* Vb = vt + ((long)b*NH + h)*DH*TPAD;
  int qrow = imin(mt*16 + ml, SEQQ - 1);
  const _Float16* Qrow = Q + (long)qrow*DH;
  v16h aq0 = load_a_contig(Qrow, 0, half);
  v16h aq1 = load_a_contig(Qrow, 32, half);
  for (int tt = wave; tt < 13; tt += 4) {          // uniform per wave
    int tc = imin(tt*16 + ml, SEQQ - 1);
    const _Float16* Krow = K + (long)tc*DH;        // B[k=e][n=t] contiguous per lane
    v8f acc = {};
    acc = wmma_f16(aq0, load_b_contig(Krow, 0,  half), acc);
    acc = wmma_f16(aq1, load_b_contig(Krow, 32, half), acc);
#pragma unroll
    for (int r = 0; r < 8; ++r) S[r + 8*half][tt*16 + ml] = acc[r] * 0.125f;
  }
  __syncthreads();
  if ((int)threadIdx.x < 16) {
    int m = threadIdx.x;
    float mx = -1e30f;
    for (int t = 0; t < SEQQ; ++t) mx = fmaxf(mx, S[m][t]);
    float sum = 0.f;
    for (int t = 0; t < SEQQ; ++t) { float e = __expf(S[m][t] - mx); S[m][t] = e; sum += e; }
    float inv = 1.f / sum;
    for (int t = 0; t < TPAD; ++t) P[m][t] = (t < SEQQ) ? (_Float16)(S[m][t] * inv) : (_Float16)0.f;
  }
  __syncthreads();
  int e0 = wave*16;
  const _Float16* Vrow = Vb + (long)(e0 + ml)*TPAD;   // B[k=t][n=e] contiguous per lane
  v8f acc = {};
  for (int k0 = 0; k0 < TPAD; k0 += 32)
    acc = wmma_f16(load_a_contig(&P[ml][0], k0, half), load_b_contig(Vrow, k0, half), acc);
#pragma unroll
  for (int r = 0; r < 8; ++r) {
    int m = mt*16 + r + 8*half;
    if (m < SEQQ) {
      long o = ((long)b*SEQQ + m)*HID + h*DH + e0 + ml;
      outb[o] = tokens[o] + acc[r];
    }
  }
}

// out2 = res + relu(x16 @ Wenc + b_enc)   (WencT: [n][k])
__global__ void __launch_bounds__(128) k_enc(const _Float16* __restrict__ A16,
    const _Float16* __restrict__ WT, const float* __restrict__ bias,
    const float* __restrict__ res, float* __restrict__ outp) {
  int b = blockIdx.x, mt = blockIdx.y, nt = blockIdx.z;
  int lane = threadIdx.x & 31, wave = threadIdx.x >> 5;
  int half = lane >> 4, ml = lane & 15;
  int rowc = imin(mt*16 + ml, SEQQ - 1);
  const _Float16* Arow = A16 + ((long)b*SEQQ + rowc)*HID;
  int n0 = nt*64 + wave*16, col = n0 + ml;
  const _Float16* Brow = WT + (long)col*HID;
  v8f acc = {};
  for (int k = 0; k < HID; k += 32)
    acc = wmma_f16(load_a_contig(Arow, k, half), load_b_contig(Brow, k, half), acc);
#pragma unroll
  for (int r = 0; r < 8; ++r) {
    int m = mt*16 + r + 8*half;
    if (m < SEQQ) {
      long o = ((long)b*SEQQ + m)*HID + col;
      outp[o] = res[o] + fmaxf(acc[r] + bias[col], 0.f);
    }
  }
}

// logits[b, n] = out2[b, 0, :] @ Wout + b_out   (WoutT: [n][k], M = 64 batch rows)
__global__ void __launch_bounds__(128) k_head(const float* __restrict__ x,
    const _Float16* __restrict__ WT, const float* __restrict__ bias, float* __restrict__ logits) {
  int mt = blockIdx.x, ng = blockIdx.y;
  int lane = threadIdx.x & 31, wave = threadIdx.x >> 5;
  int half = lane >> 4, ml = lane & 15;
  int row = mt*16 + ml;                      // batch index < 64 always
  const float* Arow = x + (long)row*SEQQ*HID;
  int n0 = ng*64 + wave*16;
  int colc = imin(n0 + ml, OUTD - 1);
  const _Float16* Brow = WT + (long)colc*HID;
  v8f acc = {};
  for (int k = 0; k < HID; k += 32)
    acc = wmma_f16(load_a32_contig(Arow, k, half), load_b_contig(Brow, k, half), acc);
#pragma unroll
  for (int r = 0; r < 8; ++r) {
    int bb = mt*16 + r + 8*half;
    int n = n0 + ml;
    if (n < OUTD) logits[(long)bb*OUTD + n] = acc[r] + bias[n];
  }
}

__global__ void k_softmax(float* __restrict__ x) {
  __shared__ float sh[256];
  int b = blockIdx.x;
  float* p = x + (long)b*OUTD;
  float mx = -1e30f;
  for (int i = threadIdx.x; i < OUTD; i += blockDim.x) mx = fmaxf(mx, p[i]);
  sh[threadIdx.x] = mx; __syncthreads();
  for (int st = 128; st > 0; st >>= 1) {
    if ((int)threadIdx.x < st) sh[threadIdx.x] = fmaxf(sh[threadIdx.x], sh[threadIdx.x + st]);
    __syncthreads();
  }
  mx = sh[0]; __syncthreads();
  float s = 0.f;
  for (int i = threadIdx.x; i < OUTD; i += blockDim.x) { float e = __expf(p[i] - mx); p[i] = e; s += e; }
  sh[threadIdx.x] = s; __syncthreads();
  for (int st = 128; st > 0; st >>= 1) {
    if ((int)threadIdx.x < st) sh[threadIdx.x] += sh[threadIdx.x + st];
    __syncthreads();
  }
  float inv = 1.f / sh[0];
  for (int i = threadIdx.x; i < OUTD; i += blockDim.x) p[i] *= inv;
}

extern "C" void kernel_launch(void* const* d_in, const int* in_sizes, int n_in,
                              void* d_out, int out_size, void* d_ws, size_t ws_size,
                              hipStream_t stream) {
  (void)in_sizes; (void)n_in; (void)out_size; (void)ws_size;
  const float* images = (const float*)d_in[0];
  const float* W_map  = (const float*)d_in[1];
  const float* b_map  = (const float*)d_in[2];
  const float* cls_t  = (const float*)d_in[3];
  const float* ln1_g  = (const float*)d_in[4];
  const float* ln1_b  = (const float*)d_in[5];
  const float* Wq     = (const float*)d_in[6];
  const float* bq     = (const float*)d_in[7];
  const float* Wk     = (const float*)d_in[8];
  const float* bk     = (const float*)d_in[9];
  const float* Wv     = (const float*)d_in[10];
  const float* bv     = (const float*)d_in[11];
  const float* ln2_g  = (const float*)d_in[12];
  const float* ln2_b  = (const float*)d_in[13];
  const float* W_enc  = (const float*)d_in[14];
  const float* b_enc  = (const float*)d_in[15];
  const float* W_out  = (const float*)d_in[16];
  const float* b_out  = (const float*)d_in[17];
  float* out = (float*)d_out;

  char* ws = (char*)d_ws;
  size_t off = 0;
  auto alloc = [&](size_t bytes) -> char* {
    off = (off + 255) & ~(size_t)255;
    char* p = ws + off; off += bytes; return p;
  };
  float*    pe     = (float*)alloc(sizeof(float)*SEQQ*HID);
  _Float16* wmapT  = (_Float16*)alloc(sizeof(_Float16)*HID*HID);
  _Float16* wencT  = (_Float16*)alloc(sizeof(_Float16)*HID*HID);
  _Float16* woutT  = (_Float16*)alloc(sizeof(_Float16)*(size_t)OUTD*HID);
  _Float16* wqT    = (_Float16*)alloc(sizeof(_Float16)*NH*DH*DH);
  _Float16* wkT    = (_Float16*)alloc(sizeof(_Float16)*NH*DH*DH);
  _Float16* wvT    = (_Float16*)alloc(sizeof(_Float16)*NH*DH*DH);
  float*    mu1    = (float*)alloc(sizeof(float)*BATCH);
  float*    rstd1  = (float*)alloc(sizeof(float)*BATCH);
  float*    mu2    = (float*)alloc(sizeof(float)*BATCH);
  float*    rstd2  = (float*)alloc(sizeof(float)*BATCH);
  float*    tokens = (float*)alloc(sizeof(float)*(size_t)BATCH*SEQQ*HID);      // later reused as out2
  float*    outb   = (float*)alloc(sizeof(float)*(size_t)BATCH*SEQQ*HID);
  _Float16* x16    = (_Float16*)alloc(sizeof(_Float16)*(size_t)BATCH*SEQQ*HID);
  _Float16* q16    = (_Float16*)alloc(sizeof(_Float16)*(size_t)BATCH*NH*SEQQ*DH);
  _Float16* k16    = (_Float16*)alloc(sizeof(_Float16)*(size_t)BATCH*NH*SEQQ*DH);
  _Float16* v16t   = (_Float16*)alloc(sizeof(_Float16)*(size_t)BATCH*NH*DH*TPAD);
  _Float16* patch16 = q16;  // alias: patches consumed before q is produced

  dim3 blk(128);
  k_pos<<<(SEQQ*HID + 255)/256, 256, 0, stream>>>(pe);
  k_cvtT<<<(HID*HID + 255)/256, 256, 0, stream>>>(W_map, wmapT, HID, HID);
  k_cvtT<<<(HID*HID + 255)/256, 256, 0, stream>>>(W_enc, wencT, HID, HID);
  k_cvtT<<<(OUTD*HID + 255)/256, 256, 0, stream>>>(W_out, woutT, HID, OUTD);
  k_cvtT_heads<<<(NH*DH*DH + 255)/256, 256, 0, stream>>>(Wq, wqT);
  k_cvtT_heads<<<(NH*DH*DH + 255)/256, 256, 0, stream>>>(Wk, wkT);
  k_cvtT_heads<<<(NH*DH*DH + 255)/256, 256, 0, stream>>>(Wv, wvT);
  k_vpad<<<(BATCH*NH*DH*(TPAD-SEQQ) + 255)/256, 256, 0, stream>>>(v16t);
  k_patches<<<(BATCH*NP*HID + 255)/256, 256, 0, stream>>>(images, patch16);
  k_cls<<<(BATCH*HID + 255)/256, 256, 0, stream>>>(cls_t, pe, tokens);
  k_embed<<<dim3(BATCH, 13, 12), blk, 0, stream>>>(patch16, wmapT, b_map, pe, tokens);
  k_stats<<<BATCH, 256, 0, stream>>>(tokens, mu1, rstd1);
  {
    long n = (long)BATCH*SEQQ*HID;
    k_ln<<<(int)((n + 255)/256), 256, 0, stream>>>(tokens, mu1, rstd1, ln1_g, ln1_b, x16);
  }
  k_qkv<<<dim3(BATCH, NH, 13), blk, 0, stream>>>(x16, wqT, wkT, wvT, bq, bk, bv, q16, k16, v16t);
  k_attn<<<dim3(BATCH, NH, 13), blk, 0, stream>>>(q16, k16, v16t, tokens, outb);
  k_stats<<<BATCH, 256, 0, stream>>>(outb, mu2, rstd2);
  {
    long n = (long)BATCH*SEQQ*HID;
    k_ln<<<(int)((n + 255)/256), 256, 0, stream>>>(outb, mu2, rstd2, ln2_g, ln2_b, x16);
  }
  k_enc<<<dim3(BATCH, 13, 12), blk, 0, stream>>>(x16, wencT, b_enc, outb, tokens);
  k_head<<<dim3(4, 16), blk, 0, stream>>>(tokens, woutT, b_out, out);
  k_softmax<<<BATCH, 256, 0, stream>>>(out);
}
